// LlamaAttention_721554505766
// MI455X (gfx1250) — matrix operand order repaired and checked by reading, not compiled, not run on hardware
//
#include <hip/hip_runtime.h>
#include <hip/hip_bf16.h>
#include <math.h>

// ---------------------------------------------------------------------------
// Problem constants (fixed by the reference harness)
// ---------------------------------------------------------------------------
#define T_TOK   4096
#define HID     4096
#define HQ      32
#define HKV     8
#define DH      128
#define SEQ     1024
#define BATCH   4
#define GQA     (HQ / HKV)           // 4
#define SCALE_Q 0.08838834764831845f // 1/sqrt(128)

typedef __attribute__((ext_vector_type(16))) __bf16        v16bf;
typedef __attribute__((ext_vector_type(8)))  __bf16        v8bf;
typedef __attribute__((ext_vector_type(8)))  float         v8f;
typedef __attribute__((ext_vector_type(4)))  unsigned int  v4u;
typedef __attribute__((ext_vector_type(8)))  int           v8i;
typedef __attribute__((ext_vector_type(4)))  int           v4i;

union BF16x16 { v16bf v; v8bf h[2]; };

#if defined(__HIP_DEVICE_COMPILE__) && __has_builtin(__builtin_amdgcn_tensor_load_to_lds) && __has_builtin(__builtin_amdgcn_s_wait_tensorcnt)
#define USE_TDM 1
#else
#define USE_TDM 0
#endif

// ---------------------------------------------------------------------------
// fp32 -> bf16 conversion (elementwise)
// ---------------------------------------------------------------------------
__global__ void cvt_f32_bf16_kernel(const float* __restrict__ src,
                                    __bf16* __restrict__ dst, int n) {
  int i = blockIdx.x * blockDim.x + threadIdx.x;
  if (i < n) dst[i] = (__bf16)src[i];
}

// ---------------------------------------------------------------------------
// bf16 GEMM: C[M,N] = A[M,K] * W[N,K]^T ; one wave -> 32(M) x 64(N) tile,
// register double-buffered so next-k fragments load while current-k WMMAs run.
// A-fragment: lane m=l&15, K chunks {sel*8..+7, 16+sel*8..+7}
// B-fragment: lane n=l&15, K chunks {sel*8..+7, 16+sel*8..+7}
// C layout  : vgpr r, lane l -> (row = r + 8*sel, col = l&15)
// ---------------------------------------------------------------------------
__global__ __launch_bounds__(32)
void gemm_bf16_wmma_kernel(const __bf16* __restrict__ A,
                           const __bf16* __restrict__ W,
                           __bf16* __restrict__ Cb,
                           float* __restrict__ Cf,
                           int M, int N, int K, int f32out) {
  const int ntile = blockIdx.x;      // N/64 tiles
  const int mtile = blockIdx.y;      // M/32 tiles
  const int lane  = threadIdx.x;
  const int n16   = lane & 15;
  const int sel   = lane >> 4;

  v8f acc[2][4];
#pragma unroll
  for (int mi = 0; mi < 2; ++mi)
#pragma unroll
    for (int j = 0; j < 4; ++j) acc[mi][j] = (v8f){};

  const __bf16* ap0 = A + (size_t)(mtile * 32 + n16) * K;
  const __bf16* ap1 = ap0 + (size_t)16 * K;
  const __bf16* wp0 = W + (size_t)(ntile * 64 + n16) * K;

  BF16x16 ac[2], bc[4];
  ac[0].h[0] = *(const v8bf*)(ap0 + sel * 8);
  ac[0].h[1] = *(const v8bf*)(ap0 + 16 + sel * 8);
  ac[1].h[0] = *(const v8bf*)(ap1 + sel * 8);
  ac[1].h[1] = *(const v8bf*)(ap1 + 16 + sel * 8);
#pragma unroll
  for (int j = 0; j < 4; ++j) {
    const __bf16* wp = wp0 + (size_t)(j * 16) * K + sel * 8;
    bc[j].h[0] = *(const v8bf*)(wp);
    bc[j].h[1] = *(const v8bf*)(wp + 16);
  }

  for (int kk = 0; kk < K; kk += 32) {
    const int kn = (kk + 32 < K) ? kk + 32 : kk;  // last iter reloads (benign)
    BF16x16 an[2], bn[4];
    an[0].h[0] = *(const v8bf*)(ap0 + kn + sel * 8);
    an[0].h[1] = *(const v8bf*)(ap0 + kn + 16 + sel * 8);
    an[1].h[0] = *(const v8bf*)(ap1 + kn + sel * 8);
    an[1].h[1] = *(const v8bf*)(ap1 + kn + 16 + sel * 8);
#pragma unroll
    for (int j = 0; j < 4; ++j) {
      const __bf16* wp = wp0 + (size_t)(j * 16) * K + kn + sel * 8;
      bn[j].h[0] = *(const v8bf*)(wp);
      bn[j].h[1] = *(const v8bf*)(wp + 16);
    }
    __builtin_prefetch(ap0 + kk + 256, 0, 1);
    __builtin_prefetch(wp0 + kk + 256, 0, 1);
#pragma unroll
    for (int j = 0; j < 4; ++j) {
      acc[0][j] = __builtin_amdgcn_wmma_f32_16x16x32_bf16(
          false, ac[0].v, false, bc[j].v, (short)0, acc[0][j], false, false);
      acc[1][j] = __builtin_amdgcn_wmma_f32_16x16x32_bf16(
          false, ac[1].v, false, bc[j].v, (short)0, acc[1][j], false, false);
    }
    ac[0] = an[0]; ac[1] = an[1];
#pragma unroll
    for (int j = 0; j < 4; ++j) bc[j] = bn[j];
  }

#pragma unroll
  for (int mi = 0; mi < 2; ++mi)
#pragma unroll
    for (int j = 0; j < 4; ++j)
#pragma unroll
      for (int r = 0; r < 8; ++r) {
        const int row = mtile * 32 + mi * 16 + r + 8 * sel;
        const int col = ntile * 64 + j * 16 + n16;
        if (f32out) Cf[(size_t)row * N + col] = acc[mi][j][r];
        else        Cb[(size_t)row * N + col] = (__bf16)acc[mi][j][r];
      }
}

// ---------------------------------------------------------------------------
// RoPE (LLaMA half-split), in place on bf16 [T, H*128]; outscale folds the
// softmax 1/sqrt(D) into Q.
// ---------------------------------------------------------------------------
__global__ void rope_kernel(__bf16* __restrict__ x, const int* __restrict__ pos,
                            int H, float outscale, int total) {
  int idx = blockIdx.x * blockDim.x + threadIdx.x;
  if (idx >= total) return;
  const int d  = idx & 63;
  const int th = idx >> 6;
  const int hh = th % H;
  const int t  = th / H;
  const float p    = (float)pos[t];
  const float freq = __expf(-(float)d * (1.0f / 64.0f) * 9.210340371976184f);
  const float ang  = p * freq;
  float s, c;
  __sincosf(ang, &s, &c);
  __bf16* base = x + (size_t)t * (H * DH) + hh * DH;
  const float x1 = (float)base[d];
  const float x2 = (float)base[d + 64];
  base[d]      = (__bf16)((x1 * c - x2 * s) * outscale);
  base[d + 64] = (__bf16)((x2 * c + x1 * s) * outscale);
}

// ---------------------------------------------------------------------------
// V transpose: V[B*S, HKV*128] -> Vt[B, HKV, 128, S]
// ---------------------------------------------------------------------------
__global__ void vtrans_kernel(const __bf16* __restrict__ V,
                              __bf16* __restrict__ Vt, int total) {
  int idx = blockIdx.x * blockDim.x + threadIdx.x;
  if (idx >= total) return;
  const int d  = idx & 127;
  const int hk = (idx >> 7) & (HKV - 1);
  const int t  = idx >> 10;
  const int b  = t / SEQ;
  const int s  = t - b * SEQ;
  Vt[((size_t)(b * HKV + hk) * DH + d) * SEQ + s] = V[idx];
}

// ---------------------------------------------------------------------------
// TDM helper: DMA a 2D tile (tile_dim0=32 contiguous elems, tile_dim1=128
// rows, row stride SEQ, 2-byte elements) from Vt into LDS.
// ---------------------------------------------------------------------------
#if USE_TDM
__device__ __forceinline__ void tdm_load_vtile(unsigned lds_addr,
                                               const __bf16* gptr) {
  const unsigned td0 = 1u << 30, td1 = 1u << 20;  // large dims: no OOB clip
  v4u g0;
  const unsigned long long ga = (unsigned long long)(uintptr_t)gptr;
  g0.x = 1u;                                  // count=1, user descriptor
  g0.y = lds_addr;                            // LDS byte address
  g0.z = (unsigned)ga;                        // global addr [31:0]
  g0.w = (unsigned)(ga >> 32) | 0x80000000u;  // addr[56:32] | type=2
  v8i g1;
  g1[0] = (int)(1u << 16);                              // data_size=1 (2B)
  g1[1] = (int)((td0 & 0xffffu) << 16);                 // tensor_dim0 lo
  g1[2] = (int)((td0 >> 16) | ((td1 & 0xffffu) << 16)); // dim0 hi | dim1 lo
  g1[3] = (int)((td1 >> 16) | (32u << 16));             // dim1 hi | tile_dim0=32
  g1[4] = (int)128u;                                    // tile_dim1=128
  g1[5] = (int)SEQ;                                     // tensor_dim0_stride lo
  g1[6] = 0;
  g1[7] = 0;
  v4i z4 = (v4i){};
#if defined(__clang_major__) && __clang_major__ >= 23
  v8i z8 = (v8i){};
  __builtin_amdgcn_tensor_load_to_lds(g0, g1, z4, z4, z8, 0);
#else
  __builtin_amdgcn_tensor_load_to_lds(g0, g1, z4, z4, 0);
#endif
}
#endif

// ---------------------------------------------------------------------------
// Flash attention, one wave per (b, h, 16-row Q tile).  Ping-pong TDM: the
// V tile for block kb+1 is DMA'd into the other LDS buffer while block kb's
// P@V WMMAs run; s_wait_tensorcnt(1) gates consumption, allowing one
// in-flight DMA across the whole iteration.
// ---------------------------------------------------------------------------
__global__ __launch_bounds__(32)
void flash_attn_wmma_kernel(const __bf16* __restrict__ Qm,
                            const __bf16* __restrict__ Km,
                            const __bf16* __restrict__ Vt,
                            __bf16* __restrict__ Ao) {
  const int qt   = blockIdx.x;   // S/16
  const int h    = blockIdx.y;   // HQ
  const int b    = blockIdx.z;   // BATCH
  const int hk   = h / GQA;
  const int lane = threadIdx.x;
  const int n16  = lane & 15;
  const int sel  = lane >> 4;

  __shared__ __attribute__((aligned(16))) __bf16 Pl[16 * 32];
#if USE_TDM
  __shared__ __attribute__((aligned(16))) __bf16 Vlds[2][128 * 32]; // 2x8KB
  const unsigned vlds_addr0 = (unsigned)(uintptr_t)&Vlds[0][0];
  const unsigned vlds_addr1 = (unsigned)(uintptr_t)&Vlds[1][0];
#endif
  const __bf16* vbase = Vt + (size_t)(b * HKV + hk) * DH * SEQ;

  const int q_hi = qt * 16 + 15;
  const int nkb  = (q_hi + 32) >> 5;

#if USE_TDM
  // Prologue DMA for block 0 (max lead time: overlaps Q-fragment loads too).
  tdm_load_vtile(vlds_addr0, vbase);
#endif

  // Resident Q fragments: 4 chunks of K=32 over D=128.
  BF16x16 qf[4];
  {
    const __bf16* qp =
        Qm + (size_t)(b * SEQ + qt * 16 + n16) * (HQ * DH) + h * DH;
#pragma unroll
    for (int c = 0; c < 4; ++c) {
      qf[c].h[0] = *(const v8bf*)(qp + c * 32 + sel * 8);
      qf[c].h[1] = *(const v8bf*)(qp + c * 32 + 16 + sel * 8);
    }
  }

  v8f o[8];
#pragma unroll
  for (int nt = 0; nt < 8; ++nt) o[nt] = (v8f){};
  float m_i[8], l_i[8];
#pragma unroll
  for (int r = 0; r < 8; ++r) { m_i[r] = -1e30f; l_i[r] = 0.0f; }

  for (int kb = 0; kb < nkb; ++kb) {
#pragma unroll
    for (int sub = 0; sub < 2; ++sub) {
      const int kbase = kb * 32 + sub * 16;
      v8f s = (v8f){};
      const __bf16* kp =
          Km + (size_t)(b * SEQ + kbase + n16) * (HKV * DH) + hk * DH;
#pragma unroll
      for (int c = 0; c < 4; ++c) {
        BF16x16 kf;
        kf.h[0] = *(const v8bf*)(kp + c * 32 + sel * 8);
        kf.h[1] = *(const v8bf*)(kp + c * 32 + 16 + sel * 8);
        s = __builtin_amdgcn_wmma_f32_16x16x32_bf16(
            false, qf[c].v, false, kf.v, (short)0, s, false, false);
      }
      const int col = kbase + n16;
#pragma unroll
      for (int r = 0; r < 8; ++r) {
        const int qrow = qt * 16 + r + 8 * sel;
        float sv = (col <= qrow) ? s[r] : -1e30f;
        float mx = sv;
        mx = fmaxf(mx, __shfl_xor(mx, 1, 32));
        mx = fmaxf(mx, __shfl_xor(mx, 2, 32));
        mx = fmaxf(mx, __shfl_xor(mx, 4, 32));
        mx = fmaxf(mx, __shfl_xor(mx, 8, 32));
        const float mnew  = fmaxf(m_i[r], mx);
        const float alpha = __expf(m_i[r] - mnew);
        const float pv    = __expf(sv - mnew);
        float rs = pv;
        rs += __shfl_xor(rs, 1, 32);
        rs += __shfl_xor(rs, 2, 32);
        rs += __shfl_xor(rs, 4, 32);
        rs += __shfl_xor(rs, 8, 32);
        l_i[r] = l_i[r] * alpha + rs;
        m_i[r] = mnew;
#pragma unroll
        for (int nt = 0; nt < 8; ++nt) o[nt][r] *= alpha;
        Pl[(r + 8 * sel) * 32 + sub * 16 + n16] = (__bf16)pv;
      }
    }
    __syncthreads();
#if USE_TDM
    // All earlier LDS reads (incl. other buffer from kb-1) are retired before
    // the next DMA may overwrite it; then kick off block kb+1's DMA.
    asm volatile("s_wait_dscnt 0x0" ::: "memory");
    if (kb + 1 < nkb)
      tdm_load_vtile(((kb + 1) & 1) ? vlds_addr1 : vlds_addr0,
                     vbase + (kb + 1) * 32);
#endif
    BF16x16 pf;
    pf.h[0] = *(const v8bf*)(&Pl[n16 * 32 + sel * 8]);
    pf.h[1] = *(const v8bf*)(&Pl[n16 * 32 + 16 + sel * 8]);
#if USE_TDM
    if (kb + 1 < nkb) __builtin_amdgcn_s_wait_tensorcnt(1); // kb's tile ready
    else              __builtin_amdgcn_s_wait_tensorcnt(0);
    const __bf16* vl = &Vlds[kb & 1][0];
#pragma unroll
    for (int nt = 0; nt < 8; ++nt) {
      BF16x16 vf;
      vf.h[0] = *(const v8bf*)(vl + (nt * 16 + n16) * 32 + sel * 8);
      vf.h[1] = *(const v8bf*)(vl + (nt * 16 + n16) * 32 + 16 + sel * 8);
      o[nt] = __builtin_amdgcn_wmma_f32_16x16x32_bf16(
          false, pf.v, false, vf.v, (short)0, o[nt], false, false);
    }
#else
#pragma unroll
    for (int nt = 0; nt < 8; ++nt) {
      const __bf16* vp = vbase + (size_t)(nt * 16 + n16) * SEQ + kb * 32;
      BF16x16 vf;
      vf.h[0] = *(const v8bf*)(vp + sel * 8);
      vf.h[1] = *(const v8bf*)(vp + 16 + sel * 8);
      o[nt] = __builtin_amdgcn_wmma_f32_16x16x32_bf16(
          false, pf.v, false, vf.v, (short)0, o[nt], false, false);
    }
#endif
    __syncthreads();
  }

#pragma unroll
  for (int r = 0; r < 8; ++r) {
    const float inv = 1.0f / l_i[r];
    __bf16* op =
        Ao + (size_t)(b * SEQ + qt * 16 + r + 8 * sel) * (HQ * DH) + h * DH;
#pragma unroll
    for (int nt = 0; nt < 8; ++nt)
      op[nt * 16 + n16] = (__bf16)(o[nt][r] * inv);
  }
}

// ---------------------------------------------------------------------------
// Launch
// ---------------------------------------------------------------------------
extern "C" void kernel_launch(void* const* d_in, const int* in_sizes, int n_in,
                              void* d_out, int out_size, void* d_ws, size_t ws_size,
                              hipStream_t stream) {
  const float* hidden    = (const float*)d_in[0];
  const int*   positions = (const int*)d_in[1];
  const float* q_w       = (const float*)d_in[2];
  const float* k_w       = (const float*)d_in[3];
  const float* v_w       = (const float*)d_in[4];
  const float* o_w       = (const float*)d_in[5];
  float*       out       = (float*)d_out;

  char* ws = (char*)d_ws;
  __bf16* h_bf  = (__bf16*)ws; ws += (size_t)T_TOK * HID * 2;
  __bf16* qw_bf = (__bf16*)ws; ws += (size_t)(HQ * DH) * HID * 2;
  __bf16* kw_bf = (__bf16*)ws; ws += (size_t)(HKV * DH) * HID * 2;
  __bf16* vw_bf = (__bf16*)ws; ws += (size_t)(HKV * DH) * HID * 2;
  __bf16* ow_bf = (__bf16*)ws; ws += (size_t)HID * (HQ * DH) * 2;
  __bf16* Qb    = (__bf16*)ws; ws += (size_t)T_TOK * (HQ * DH) * 2;
  __bf16* Kb    = (__bf16*)ws; ws += (size_t)T_TOK * (HKV * DH) * 2;
  __bf16* Vb    = (__bf16*)ws; ws += (size_t)T_TOK * (HKV * DH) * 2;
  __bf16* Vt    = (__bf16*)ws; ws += (size_t)T_TOK * (HKV * DH) * 2;
  __bf16* attn  = h_bf; // hidden_bf is dead after the QKV GEMMs

  const int thr = 256;
  { // fp32 -> bf16
    int n = T_TOK * HID;
    cvt_f32_bf16_kernel<<<(n + thr - 1) / thr, thr, 0, stream>>>(hidden, h_bf, n);
    n = HQ * DH * HID;
    cvt_f32_bf16_kernel<<<(n + thr - 1) / thr, thr, 0, stream>>>(q_w, qw_bf, n);
    n = HKV * DH * HID;
    cvt_f32_bf16_kernel<<<(n + thr - 1) / thr, thr, 0, stream>>>(k_w, kw_bf, n);
    cvt_f32_bf16_kernel<<<(n + thr - 1) / thr, thr, 0, stream>>>(v_w, vw_bf, n);
    n = HID * HQ * DH;
    cvt_f32_bf16_kernel<<<(n + thr - 1) / thr, thr, 0, stream>>>(o_w, ow_bf, n);
  }
  { // QKV projections
    dim3 gq((HQ * DH) / 64, T_TOK / 32);
    gemm_bf16_wmma_kernel<<<gq, 32, 0, stream>>>(h_bf, qw_bf, Qb, nullptr,
                                                 T_TOK, HQ * DH, HID, 0);
    dim3 gk((HKV * DH) / 64, T_TOK / 32);
    gemm_bf16_wmma_kernel<<<gk, 32, 0, stream>>>(h_bf, kw_bf, Kb, nullptr,
                                                 T_TOK, HKV * DH, HID, 0);
    gemm_bf16_wmma_kernel<<<gk, 32, 0, stream>>>(h_bf, vw_bf, Vb, nullptr,
                                                 T_TOK, HKV * DH, HID, 0);
  }
  { // RoPE
    int nq = T_TOK * HQ * 64;
    rope_kernel<<<(nq + thr - 1) / thr, thr, 0, stream>>>(Qb, positions, HQ,
                                                          SCALE_Q, nq);
    int nk = T_TOK * HKV * 64;
    rope_kernel<<<(nk + thr - 1) / thr, thr, 0, stream>>>(Kb, positions, HKV,
                                                          1.0f, nk);
  }
  { // V transpose
    int n = T_TOK * HKV * DH;
    vtrans_kernel<<<(n + thr - 1) / thr, thr, 0, stream>>>(Vb, Vt, n);
  }
  { // flash attention
    dim3 ga(SEQ / 16, HQ, BATCH);
    flash_attn_wmma_kernel<<<ga, 32, 0, stream>>>(Qb, Kb, Vt, attn);
  }
  { // output projection (f32)
    dim3 go(HID / 64, T_TOK / 32);
    gemm_bf16_wmma_kernel<<<go, 32, 0, stream>>>(attn, ow_bf, nullptr, out,
                                                 T_TOK, HID, HQ * DH, 1);
  }
}
